// Encoder_88055419502798
// MI455X (gfx1250) — matrix operand order repaired and checked by reading, not compiled, not run on hardware
//
#include <hip/hip_runtime.h>
#include <hip/hip_bf16.h>
#include <math.h>

// Problem constants (from reference)
#define LL   6
#define CC   192
#define FCH_ 768
#define KK   3
#define HH   2
#define WW   10
#define BB   8
#define TT   768
#define KC_  96
#define NR   21   // 2W+1 relative buckets

typedef __attribute__((ext_vector_type(16))) _Float16 v16h;
typedef __attribute__((ext_vector_type(8)))  _Float16 h8;
typedef __attribute__((ext_vector_type(8)))  float    v8f;
typedef __attribute__((ext_vector_type(4)))  int      v4i;

#define WMMA_F16(A, B, C) \
  __builtin_amdgcn_wmma_f32_16x16x32_f16(false, (A), false, (B), (short)0, (C), false, false)

// ---------- async 16B global->LDS copy with safe fallback ----------
#if defined(__HIP_DEVICE_COMPILE__) && __has_builtin(__builtin_amdgcn_global_load_async_to_lds_b128)
#define HAVE_ASYNC_LDS 1
#else
#define HAVE_ASYNC_LDS 0
#endif

#if HAVE_ASYNC_LDS
typedef __attribute__((address_space(1))) v4i* gv4i_p;   // global
typedef __attribute__((address_space(3))) v4i* lv4i_p;   // LDS
#endif

__device__ __forceinline__ void async_cp16(_Float16* dst, const _Float16* src) {
#if HAVE_ASYNC_LDS
  __builtin_amdgcn_global_load_async_to_lds_b128((gv4i_p)src, (lv4i_p)dst, 0, 0);
#else
  *(float4*)dst = *(const float4*)src;
#endif
}
template <int N>
__device__ __forceinline__ void async_wait_c() {
#if HAVE_ASYNC_LDS
#if __has_builtin(__builtin_amdgcn_s_wait_asynccnt)
  __builtin_amdgcn_s_wait_asynccnt(N);
#else
  asm volatile("s_wait_asynccnt %0" :: "i"(N) : "memory");
#endif
#endif
}

// ---------- fragment loaders: contiguous 2x16B LDS reads ----------
__device__ __forceinline__ v16h a_frag(const _Float16* tile, int stride) {
  int lane = threadIdx.x & 31;
  int m  = lane & 15;
  int k0 = (lane >> 4) << 3;     // 0 or 8
  const _Float16* p = tile + m * stride + k0;
  h8 lo = *(const h8*)(p);
  h8 hi = *(const h8*)(p + 16);
  return __builtin_shufflevector(lo, hi, 0,1,2,3,4,5,6,7,8,9,10,11,12,13,14,15);
}
__device__ __forceinline__ v16h b_frag(const _Float16* tile, int stride) {
  int lane = threadIdx.x & 31;
  int n  = lane & 15;
  int kb = (lane >> 4) << 3;     // 0 or 8
  const _Float16* p = tile + n * stride + kb;
  h8 lo = *(const h8*)(p);
  h8 hi = *(const h8*)(p + 16);
  return __builtin_shufflevector(lo, hi, 0,1,2,3,4,5,6,7,8,9,10,11,12,13,14,15);
}

// ---------- time-major GEMM (double-buffered, pipelined async staging) ----------
// Y[b][t][n] = sum_c A[b][t][c] * W[n][c] + bias[n]
__global__ __launch_bounds__(128) void k_gemm_tm(
    const _Float16* __restrict__ Ah,   // [B][T][Kd]
    const _Float16* __restrict__ Bh,   // [N][Kd]
    const float* __restrict__ bias,    // [N]
    float* __restrict__ Yf,            // [B][T][N] or null
    _Float16* __restrict__ Yh,         // [B][T][N] or null
    const float* __restrict__ omask,   // [B][T] or null
    int Kd, int Nch, int T, int relu)
{
  int t0 = blockIdx.x * 64;
  int n0 = blockIdx.y * 64;
  int b  = blockIdx.z;
  const _Float16* Ab = Ah + (size_t)b * T * Kd;

  __shared__ __align__(16) _Float16 sA[2][64 * 32];
  __shared__ __align__(16) _Float16 sB[2][64 * 32];

  int tid = threadIdx.x, w = tid >> 5, lane = tid & 31;
  int n = lane & 15, hi = lane >> 4;

  v8f acc[4];
#pragma unroll
  for (int s = 0; s < 4; ++s)
#pragma unroll
    for (int i = 0; i < 8; ++i) acc[s][i] = 0.f;

  auto issue = [&](int buf, int k0) {
#pragma unroll
    for (int j = 0; j < 2; ++j) {        // 4 async ops per thread per tile
      int ch = tid + j * 128;
      int r = ch >> 2, c8 = (ch & 3) << 3;
      async_cp16(&sA[buf][r * 32 + c8], Ab + (size_t)(t0 + r) * Kd + k0 + c8);
      async_cp16(&sB[buf][r * 32 + c8], Bh + (size_t)(n0 + r) * Kd + k0 + c8);
    }
  };

  int nk = Kd >> 5;
  issue(0, 0);
  for (int kk = 0; kk < nk; ++kk) {
    int cur = kk & 1;
    if (kk + 1 < nk) { issue(cur ^ 1, (kk + 1) << 5); async_wait_c<4>(); }
    else             { async_wait_c<0>(); }
    __syncthreads();
    v16h a = a_frag(&sA[cur][(w * 16) * 32], 32);
#pragma unroll
    for (int ss = 0; ss < 4; ++ss) {
      v16h bf = b_frag(&sB[cur][(ss * 16) * 32], 32);
      acc[ss] = WMMA_F16(a, bf, acc[ss]);
    }
    __syncthreads();
  }

  const float* mk = omask ? omask + (size_t)b * T : nullptr;
#pragma unroll
  for (int ss = 0; ss < 4; ++ss)
#pragma unroll
    for (int v = 0; v < 8; ++v) {
      int t = t0 + w * 16 + v + 8 * hi;
      int m = n0 + ss * 16 + n;
      float val = acc[ss][v] + bias[m];
      if (relu) val = fmaxf(val, 0.f);
      if (mk) val *= mk[t];
      size_t idx = (size_t)b * T * Nch + (size_t)t * Nch + m;
      if (Yf) Yf[idx] = val;
      if (Yh) Yh[idx] = (_Float16)val;
    }
}

// ---------- K=3 conv as GEMM (double-buffered); weights pre-reordered to [N][k][Cin] ----------
__global__ __launch_bounds__(128) void k_gemm_conv_tm(
    const _Float16* __restrict__ Ah,   // [B][T][Cin]
    const _Float16* __restrict__ Bh,   // [N][3*Cin] (k-major)
    const float* __restrict__ bias,
    float* __restrict__ Yf, _Float16* __restrict__ Yh,
    const float* __restrict__ omask,
    int Cin, int Nch, int T, int relu)
{
  int t0 = blockIdx.x * 64;
  int n0 = blockIdx.y * 64;
  int b  = blockIdx.z;
  int K3 = Cin * 3;
  const _Float16* Ab = Ah + (size_t)b * T * Cin;

  __shared__ __align__(16) _Float16 sA[2][64 * 32];
  __shared__ __align__(16) _Float16 sB[2][64 * 32];

  int tid = threadIdx.x, w = tid >> 5, lane = tid & 31;
  int n = lane & 15, hi = lane >> 4;

  v8f acc[4];
#pragma unroll
  for (int s = 0; s < 4; ++s)
#pragma unroll
    for (int i = 0; i < 8; ++i) acc[s][i] = 0.f;

  // Always issue exactly 4 async ops per thread (OOB rows clamped, patched to zero
  // after the wait) so the pipelined s_wait_asynccnt<=4 is sound.
  auto issue = [&](int buf, int k0, int& oob) {
    int sh  = k0 / Cin;            // Cin multiple of 32: chunk stays in one shift
    int kc0 = k0 - sh * Cin;
    oob = 0;
#pragma unroll
    for (int j = 0; j < 2; ++j) {
      int ch = tid + j * 128;
      int r = ch >> 2, c8 = (ch & 3) << 3;
      int ts  = t0 + r + sh - 1;
      int tsc = ts < 0 ? 0 : (ts >= T ? T - 1 : ts);
      if (ts != tsc) oob |= (1 << j);
      async_cp16(&sA[buf][r * 32 + c8], Ab + (size_t)tsc * Cin + kc0 + c8);
      async_cp16(&sB[buf][r * 32 + c8], Bh + (size_t)(n0 + r) * K3 + k0 + c8);
    }
  };

  int nk = K3 >> 5;
  int oob_cur = 0, oob_nxt = 0;
  issue(0, 0, oob_cur);
  for (int kk = 0; kk < nk; ++kk) {
    int cur = kk & 1;
    if (kk + 1 < nk) { issue(cur ^ 1, (kk + 1) << 5, oob_nxt); async_wait_c<4>(); }
    else             { oob_nxt = 0; async_wait_c<0>(); }
    if (oob_cur) {                     // zero-pad patch after async writes landed
#pragma unroll
      for (int j = 0; j < 2; ++j)
        if (oob_cur & (1 << j)) {
          int ch = tid + j * 128;
          int r = ch >> 2, c8 = (ch & 3) << 3;
          float4 z = {0.f, 0.f, 0.f, 0.f};
          *(float4*)&sA[cur][r * 32 + c8] = z;
        }
    }
    __syncthreads();
    v16h a = a_frag(&sA[cur][(w * 16) * 32], 32);
#pragma unroll
    for (int ss = 0; ss < 4; ++ss) {
      v16h bf = b_frag(&sB[cur][(ss * 16) * 32], 32);
      acc[ss] = WMMA_F16(a, bf, acc[ss]);
    }
    __syncthreads();
    oob_cur = oob_nxt;
  }

  const float* mk = omask ? omask + (size_t)b * T : nullptr;
#pragma unroll
  for (int ss = 0; ss < 4; ++ss)
#pragma unroll
    for (int v = 0; v < 8; ++v) {
      int t = t0 + w * 16 + v + 8 * hi;
      int m = n0 + ss * 16 + n;
      float val = acc[ss][v] + bias[m];
      if (relu) val = fmaxf(val, 0.f);
      if (mk) val *= mk[t];
      size_t idx = (size_t)b * T * Nch + (size_t)t * Nch + m;
      if (Yf) Yf[idx] = val;
      if (Yh) Yh[idx] = (_Float16)val;
    }
}

// ---------- flash attention (time-major heads) ----------
// Qh/Kh/Vh/Oh: [B][T][C] f16, head h = columns [h*96, h*96+96)
__global__ __launch_bounds__(128) void k_attn_tm(
    const _Float16* __restrict__ Qh, const _Float16* __restrict__ Kh,
    const _Float16* __restrict__ Vh,
    const float* __restrict__ erk, const float* __restrict__ erv,  // [21][96]
    _Float16* __restrict__ Oh, int T)
{
  int t0 = blockIdx.x * 64;
  int h  = blockIdx.y;
  int b  = blockIdx.z;
  int hb = h * KC_;
  const _Float16* Qb = Qh + (size_t)b * T * CC;
  const _Float16* Kb = Kh + (size_t)b * T * CC;
  const _Float16* Vb = Vh + (size_t)b * T * CC;
  _Float16*       Ob = Oh + (size_t)b * T * CC;

  __shared__ __align__(16) _Float16 kt[64 * 96];   // [s_local][d]
  __shared__ __align__(16) _Float16 vtT[96 * 64];  // [d][s_local]
  __shared__ float qrel[64 * NR];
  __shared__ float rsum[4][16 * NR];
  __shared__ __align__(16) _Float16 prow[4][16 * 32];

  int tid = threadIdx.x, w = tid >> 5, lane = tid & 31;
  int n = lane & 15, hi = lane >> 4;
  const float scale = rsqrtf((float)KC_);

  for (int e = tid; e < 64 * NR; e += 128) {
    int tl = e / NR, r = e % NR;
    const _Float16* qr = Qb + (size_t)(t0 + tl) * CC + hb;
    float s = 0.f;
    for (int d = 0; d < KC_; ++d) s += (float)qr[d] * erk[r * KC_ + d];
    qrel[e] = s * scale;
  }
  for (int e = tid; e < 4 * 16 * NR; e += 128) (&rsum[0][0])[e] = 0.f;

  v16h qa[3];
  {
    int m  = lane & 15;
    int kg = (lane >> 4) << 3;
    const _Float16* qp = Qb + (size_t)(t0 + w * 16 + m) * CC + hb;
#pragma unroll
    for (int kc = 0; kc < 3; ++kc) {
      h8 lo = *(const h8*)(qp + kc * 32 + kg);
      h8 hic = *(const h8*)(qp + kc * 32 + kg + 16);
      qa[kc] = __builtin_shufflevector(lo, hic, 0,1,2,3,4,5,6,7,8,9,10,11,12,13,14,15);
    }
  }

  float m_run[8], l_run[8];
  v8f oacc[6];
#pragma unroll
  for (int i = 0; i < 8; ++i) { m_run[i] = -1e30f; l_run[i] = 0.f; }
#pragma unroll
  for (int dt = 0; dt < 6; ++dt)
#pragma unroll
    for (int i = 0; i < 8; ++i) oacc[dt][i] = 0.f;

  __syncthreads();

  for (int s0 = 0; s0 < T; s0 += 64) {
    __syncthreads();
    // K tile: async linear copies overlap with V transpose below
#pragma unroll
    for (int j = 0; j < 6; ++j) {
      int ch = tid + j * 128;
      int r = ch / 12, c8 = (ch % 12) << 3;
      async_cp16(&kt[r * 96 + c8], Kb + (size_t)(s0 + r) * CC + hb + c8);
    }
#pragma unroll
    for (int j = 0; j < 6; ++j) {
      int ch = tid + j * 128;
      int sl = ch / 12, d8 = (ch % 12) << 3;
      h8 vv = *(const h8*)(Vb + (size_t)(s0 + sl) * CC + hb + d8);
#pragma unroll
      for (int i = 0; i < 8; ++i) vtT[(d8 + i) * 64 + sl] = vv[i];
    }
    async_wait_c<0>();
    __syncthreads();

#pragma unroll
    for (int sc = 0; sc < 2; ++sc) {
      v8f sf[2];
#pragma unroll
      for (int half = 0; half < 2; ++half) {
        int ss = sc * 2 + half;
        v8f s8;
#pragma unroll
        for (int i = 0; i < 8; ++i) s8[i] = 0.f;
#pragma unroll
        for (int kc = 0; kc < 3; ++kc) {
          v16h bf = b_frag(&kt[(ss * 16) * 96 + kc * 32], 96);
          s8 = WMMA_F16(qa[kc], bf, s8);
        }
#pragma unroll
        for (int vv = 0; vv < 8; ++vv) {
          int tl = w * 16 + vv + 8 * hi;
          int r  = (t0 + tl) - (s0 + ss * 16 + n) + WW;
          r = r < 0 ? 0 : (r > 2 * WW ? 2 * WW : r);
          s8[vv] = s8[vv] * scale + qrel[tl * NR + r];
        }
        sf[half] = s8;
      }
      float mt[8];
#pragma unroll
      for (int vv = 0; vv < 8; ++vv) {
        float x = fmaxf(sf[0][vv], sf[1][vv]);
        x = fmaxf(x, __shfl_xor(x, 1, 16));
        x = fmaxf(x, __shfl_xor(x, 2, 16));
        x = fmaxf(x, __shfl_xor(x, 4, 16));
        x = fmaxf(x, __shfl_xor(x, 8, 16));
        mt[vv] = x;
      }
      float alpha[8];
#pragma unroll
      for (int vv = 0; vv < 8; ++vv) {
        float mn = fmaxf(m_run[vv], mt[vv]);
        alpha[vv] = __expf(m_run[vv] - mn);
        m_run[vv] = mn;
      }
#pragma unroll
      for (int dt = 0; dt < 6; ++dt)
#pragma unroll
        for (int vv = 0; vv < 8; ++vv) oacc[dt][vv] *= alpha[vv];
#pragma unroll
      for (int vv = 0; vv < 8; ++vv) {
        int lr = vv + 8 * hi;
        rsum[w][lr * NR + n] *= alpha[vv];
        if (n < NR - 16) rsum[w][lr * NR + 16 + n] *= alpha[vv];
      }
      float rs[8];
#pragma unroll
      for (int vv = 0; vv < 8; ++vv) rs[vv] = 0.f;
#pragma unroll
      for (int half = 0; half < 2; ++half) {
#pragma unroll
        for (int vv = 0; vv < 8; ++vv) {
          float p = __expf(sf[half][vv] - m_run[vv]);
          rs[vv] += p;
          int lr = vv + 8 * hi;
          int r  = (t0 + w * 16 + lr) - (s0 + (sc * 2 + half) * 16 + n) + WW;
          r = r < 0 ? 0 : (r > 2 * WW ? 2 * WW : r);
          atomicAdd(&rsum[w][lr * NR + r], p);
          prow[w][lr * 32 + half * 16 + n] = (_Float16)p;
        }
      }
#pragma unroll
      for (int vv = 0; vv < 8; ++vv) {
        float x = rs[vv];
        x += __shfl_xor(x, 1, 16);
        x += __shfl_xor(x, 2, 16);
        x += __shfl_xor(x, 4, 16);
        x += __shfl_xor(x, 8, 16);
        l_run[vv] = l_run[vv] * alpha[vv] + x;
      }
      __syncthreads();
      v16h pa = a_frag(&prow[w][0], 32);
#pragma unroll
      for (int dt = 0; dt < 6; ++dt) {
        v16h bf = b_frag(&vtT[(dt * 16) * 64 + sc * 32], 64);
        oacc[dt] = WMMA_F16(pa, bf, oacc[dt]);
      }
      __syncthreads();
    }
  }
  __syncthreads();
#pragma unroll
  for (int dt = 0; dt < 6; ++dt) {
#pragma unroll
    for (int vv = 0; vv < 8; ++vv) {
      int lr = vv + 8 * hi;
      int t  = t0 + w * 16 + lr;
      int d  = dt * 16 + n;
      float rel = 0.f;
#pragma unroll
      for (int r = 0; r < NR; ++r) rel += rsum[w][lr * NR + r] * erv[r * KC_ + d];
      Ob[(size_t)t * CC + hb + d] = (_Float16)((oacc[dt][vv] + rel) / l_run[vv]);
    }
  }
}

// ---------- residual + LayerNorm, wave-per-row ----------
__global__ __launch_bounds__(256) void k_add_ln_tm(
    float* __restrict__ xc, const float* __restrict__ yb,
    const float* __restrict__ g, const float* __restrict__ be,
    const float* __restrict__ mask, _Float16* __restrict__ xh)
{
  int row  = blockIdx.x * 8 + (threadIdx.x >> 5);   // b*T + t
  int lane = threadIdx.x & 31;
  float* xr = xc + (size_t)row * CC;
  const float* yr = yb + (size_t)row * CC;
  float vbuf[6];
  float s = 0.f, s2 = 0.f;
#pragma unroll
  for (int i = 0; i < 6; ++i) {
    float v = xr[lane + i * 32] + yr[lane + i * 32];
    vbuf[i] = v; s += v; s2 += v * v;
  }
#pragma unroll
  for (int m = 1; m < 32; m <<= 1) { s += __shfl_xor(s, m, 32); s2 += __shfl_xor(s2, m, 32); }
  float mean = s / CC;
  float inv  = rsqrtf(s2 / CC - mean * mean + 1e-5f);
  float mk = mask[row];
#pragma unroll
  for (int i = 0; i < 6; ++i) {
    int c = lane + i * 32;
    float o = g[c] * (vbuf[i] - mean) * inv + be[c];
    xr[c] = o;
    xh[(size_t)row * CC + c] = (_Float16)(o * mk);
  }
}

// ---------- conversions / transposes ----------
__global__ void k_cvt(_Float16* __restrict__ o, const float* __restrict__ i, int n) {
  int e = blockIdx.x * blockDim.x + threadIdx.x;
  if (e < n) o[e] = (_Float16)i[e];
}
__global__ void k_cvt_conv(_Float16* __restrict__ o, const float* __restrict__ in,
                           int rows, int Cin) {
  int e = blockIdx.x * blockDim.x + threadIdx.x;
  int n = rows * Cin * 3;
  if (e >= n) return;
  int row = e / (Cin * 3);
  int r2  = e % (Cin * 3);
  int c = r2 / 3, k = r2 % 3;
  o[(size_t)row * Cin * 3 + k * Cin + c] = (_Float16)in[e];
}
__global__ void k_in_tr(float* __restrict__ xc, _Float16* __restrict__ xh,
                        const float* __restrict__ x, const float* __restrict__ mask) {
  int e = blockIdx.x * blockDim.x + threadIdx.x;
  int n = BB * TT * CC;
  if (e >= n) return;
  int b = e / (TT * CC);
  int r = e % (TT * CC);
  int t = r / CC, c = r % CC;
  float v = x[(size_t)b * CC * TT + (size_t)c * TT + t];
  xc[e] = v;
  xh[e] = (_Float16)(v * mask[(size_t)b * TT + t]);
}
__global__ void k_out_tr(float* __restrict__ out, const float* __restrict__ xc,
                         const float* __restrict__ mask) {
  int e = blockIdx.x * blockDim.x + threadIdx.x;
  int n = BB * CC * TT;
  if (e >= n) return;
  int b = e / (CC * TT);
  int r = e % (CC * TT);
  int c = r / TT, t = r % TT;
  out[e] = xc[(size_t)b * TT * CC + (size_t)t * CC + c] * mask[(size_t)b * TT + t];
}

extern "C" void kernel_launch(void* const* d_in, const int* in_sizes, int n_in,
                              void* d_out, int out_size, void* d_ws, size_t ws_size,
                              hipStream_t stream) {
  const float* x    = (const float*)d_in[0];
  const float* mask = (const float*)d_in[1];
  const float* w_q  = (const float*)d_in[2];
  const float* b_q  = (const float*)d_in[3];
  const float* w_k  = (const float*)d_in[4];
  const float* b_k  = (const float*)d_in[5];
  const float* w_v  = (const float*)d_in[6];
  const float* b_v  = (const float*)d_in[7];
  const float* w_o  = (const float*)d_in[8];
  const float* b_o  = (const float*)d_in[9];
  const float* erk  = (const float*)d_in[10];
  const float* erv  = (const float*)d_in[11];
  const float* g1   = (const float*)d_in[12];
  const float* be1  = (const float*)d_in[13];
  const float* w_f1 = (const float*)d_in[14];
  const float* b_f1 = (const float*)d_in[15];
  const float* w_f2 = (const float*)d_in[16];
  const float* b_f2 = (const float*)d_in[17];
  const float* g2   = (const float*)d_in[18];
  const float* be2  = (const float*)d_in[19];
  (void)in_sizes; (void)n_in; (void)ws_size;

  const size_t NTC = (size_t)BB * TT * CC;
  const size_t NTF = (size_t)BB * TT * FCH_;
  const size_t WPW = (size_t)LL * CC * CC;
  const size_t WC1 = (size_t)LL * FCH_ * CC * KK;
  const size_t WC2 = (size_t)LL * CC * FCH_ * KK;

  char* p = (char*)d_ws;
  float*     xc  = (float*)p;      p += NTC * 4;
  float*     yb  = (float*)p;      p += NTC * 4;
  _Float16*  xch = (_Float16*)p;   p += NTC * 2;
  _Float16*  qh  = (_Float16*)p;   p += NTC * 2;
  _Float16*  kh  = (_Float16*)p;   p += NTC * 2;
  _Float16*  vh  = (_Float16*)p;   p += NTC * 2;
  _Float16*  abh = (_Float16*)p;   p += NTC * 2;
  _Float16*  hh  = (_Float16*)p;   p += NTF * 2;
  _Float16*  wqh = (_Float16*)p;   p += WPW * 2;
  _Float16*  wkh = (_Float16*)p;   p += WPW * 2;
  _Float16*  wvh = (_Float16*)p;   p += WPW * 2;
  _Float16*  woh = (_Float16*)p;   p += WPW * 2;
  _Float16*  wf1h = (_Float16*)p;  p += WC1 * 2;
  _Float16*  wf2h = (_Float16*)p;  p += WC2 * 2;

  k_cvt<<<((int)WPW + 255) / 256, 256, 0, stream>>>(wqh, w_q, (int)WPW);
  k_cvt<<<((int)WPW + 255) / 256, 256, 0, stream>>>(wkh, w_k, (int)WPW);
  k_cvt<<<((int)WPW + 255) / 256, 256, 0, stream>>>(wvh, w_v, (int)WPW);
  k_cvt<<<((int)WPW + 255) / 256, 256, 0, stream>>>(woh, w_o, (int)WPW);
  k_cvt_conv<<<((int)WC1 + 255) / 256, 256, 0, stream>>>(wf1h, w_f1, LL * FCH_, CC);
  k_cvt_conv<<<((int)WC2 + 255) / 256, 256, 0, stream>>>(wf2h, w_f2, LL * CC, FCH_);

  int N3 = (int)NTC;
  k_in_tr<<<(N3 + 255) / 256, 256, 0, stream>>>(xc, xch, x, mask);

  dim3 blk(128);
  dim3 g_pw(TT / 64, CC / 64, BB);
  dim3 g_c1(TT / 64, FCH_ / 64, BB);
  dim3 g_c2(TT / 64, CC / 64, BB);
  dim3 g_at(TT / 64, HH, BB);
  int lnB = BB * TT / 8;

  for (int l = 0; l < LL; ++l) {
    const _Float16* wq = wqh + (size_t)l * CC * CC;
    const _Float16* wk = wkh + (size_t)l * CC * CC;
    const _Float16* wv = wvh + (size_t)l * CC * CC;
    const _Float16* wo = woh + (size_t)l * CC * CC;
    const float* bq = b_q + (size_t)l * CC;
    const float* bk = b_k + (size_t)l * CC;
    const float* bv = b_v + (size_t)l * CC;
    const float* bo = b_o + (size_t)l * CC;
    const float* ek = erk + (size_t)l * NR * KC_;
    const float* ev = erv + (size_t)l * NR * KC_;
    const _Float16* wf1 = wf1h + (size_t)l * FCH_ * CC * KK;
    const float*    bf1 = b_f1 + (size_t)l * FCH_;
    const _Float16* wf2 = wf2h + (size_t)l * CC * FCH_ * KK;
    const float*    bf2 = b_f2 + (size_t)l * CC;

    k_gemm_tm<<<g_pw, blk, 0, stream>>>(xch, wq, bq, nullptr, qh, nullptr, CC, CC, TT, 0);
    k_gemm_tm<<<g_pw, blk, 0, stream>>>(xch, wk, bk, nullptr, kh, nullptr, CC, CC, TT, 0);
    k_gemm_tm<<<g_pw, blk, 0, stream>>>(xch, wv, bv, nullptr, vh, nullptr, CC, CC, TT, 0);
    k_attn_tm<<<g_at, blk, 0, stream>>>(qh, kh, vh, ek, ev, abh, TT);
    k_gemm_tm<<<g_pw, blk, 0, stream>>>(abh, wo, bo, yb, nullptr, nullptr, CC, CC, TT, 0);
    k_add_ln_tm<<<lnB, 256, 0, stream>>>(xc, yb, g1 + (size_t)l * CC, be1 + (size_t)l * CC, mask, xch);
    k_gemm_conv_tm<<<g_c1, blk, 0, stream>>>(xch, wf1, bf1, nullptr, hh, mask, CC, FCH_, TT, 1);
    k_gemm_conv_tm<<<g_c2, blk, 0, stream>>>(hh, wf2, bf2, yb, nullptr, mask, FCH_, CC, TT, 0);
    k_add_ln_tm<<<lnB, 256, 0, stream>>>(xc, yb, g2 + (size_t)l * CC, be2 + (size_t)l * CC, mask, xch);
  }
  k_out_tr<<<(N3 + 255) / 256, 256, 0, stream>>>((float*)d_out, xc, mask);
}
